// VectorQuantizer_60997125538172
// MI455X (gfx1250) — matrix operand order repaired and checked by reading, not compiled, not run on hardware
//
#include <hip/hip_runtime.h>
#include <hip/hip_bf16.h>
#include <stdint.h>

// ---------------- problem constants (match reference) ----------------
#define KENT   512          // codebook entries
#define DIM    64           // embedding dim
#define HW     4096         // 64*64 spatial
#define BATCH  32
#define NVEC   (BATCH*HW)   // 131072 vectors
#define NBLOCKS 512         // 8 waves/block * 2 tiles/wave * 512 = 8192 tiles
#define BPRE_UINTS (32*2*32*8)  // jt(32) * kt(2) * lane(32) * vgpr(8) = 16384

typedef __attribute__((ext_vector_type(16))) __bf16       v16bf;
typedef __attribute__((ext_vector_type(8)))  float        v8f;
typedef __attribute__((ext_vector_type(8)))  unsigned int v8u;

// round-to-nearest-even float -> bf16 (bit pattern)
__device__ __forceinline__ unsigned int f2bf(float f) {
  unsigned int u = __float_as_uint(f);
  u += 0x7FFFu + ((u >> 16) & 1u);
  return (u >> 16) & 0xFFFFu;
}

__device__ __forceinline__ v8f wmma_bf16(v16bf a, v16bf b, v8f c) {
  return __builtin_amdgcn_wmma_f32_16x16x32_bf16(false, a, false, b,
                                                 (short)0, c, false, false);
}

// ---------------------------------------------------------------------
// Prep: build the B operand image (codebook^T tiles) in the exact WMMA
// per-lane VGPR layout, plus per-entry squared norms.
//   B layout (16x16x32 bf16, 32x16 B): col=lane%16, h=lane/16,
//   K = ((v<4)?0:16) + 8*h + 2*(v&3) (+1).  Element index:
//   Bpre[((jt*2+kt)*32 + lane)*8 + v]
// ---------------------------------------------------------------------
__global__ __launch_bounds__(256)
void vq_prep(const float* __restrict__ cb,
             unsigned int* __restrict__ Bpre,
             float* __restrict__ cnorm) {
  int t = blockIdx.x * 256 + threadIdx.x;     // grid 64*256 = 16384
  int v    = t & 7;
  int lane = (t >> 3) & 31;
  int kt   = (t >> 8) & 1;
  int jt   = t >> 9;
  int j = jt * 16 + (lane & 15);
  int k = kt * 32 + ((v < 4) ? 0 : 16) + 8 * (lane >> 4) + 2 * (v & 3);
  float x0 = cb[j * DIM + k];
  float x1 = cb[j * DIM + k + 1];
  Bpre[t] = f2bf(x0) | (f2bf(x1) << 16);

  if (t < KENT) {
    float s = 0.f;
    #pragma unroll 8
    for (int d = 0; d < DIM; ++d) { float c = cb[t * DIM + d]; s += c * c; }
    cnorm[t] = s;
  }
}

// ---------------------------------------------------------------------
// Main: one wave = two adjacent 16-row tiles (32 vectors) vs all 512
// entries; every B tile load feeds 4 WMMAs.  B and cnorm are software
// double-buffered so next-tile loads overlap current-tile WMMAs.
// dist_j = ||c_j||^2 - 2 * z . c_j   (||z||^2 constant per row, dropped)
// ---------------------------------------------------------------------
__global__ __launch_bounds__(256)
void vq_main(const float* __restrict__ z,
             const float* __restrict__ cb,
             const unsigned int* __restrict__ Bpre,
             const float* __restrict__ cnorm,
             float* __restrict__ out,
             float* __restrict__ partial) {
  __shared__ int   idxS[8][2][16];
  __shared__ float lossS[8];

  const int tid  = threadIdx.x;
  const int lane = tid & 31;
  const int lm   = lane & 15;     // column slot / row slot
  const int h    = lane >> 4;     // half of wave
  const int warp = tid >> 5;

  const int tpair = blockIdx.x * 8 + warp;     // 0..4095, two tiles each
  const int n0    = tpair << 5;                // 32 rows
  const int b     = n0 >> 12;                  // / 4096 (pair never crosses b)
  const int hw0   = n0 & 4095;
  const float* zb = z + ((size_t)b * DIM * HW + hw0);

  // ---- build A (2 tiles x 16x64 bf16) : ISA 16-bit A layout ----
  // lane half h, VGPR v: K = ((v<4)?0:16) + 8*h + 2*(v&3)  (+32 for k-step 1)
  v16bf a[2][2];
  #pragma unroll
  for (int m = 0; m < 2; ++m) {
    v8u u0, u1;
    #pragma unroll
    for (int v = 0; v < 8; ++v) {
      int kb = ((v < 4) ? 0 : 16) + 8 * h + 2 * (v & 3);
      int row = m * 16 + lm;
      float x0 = zb[(size_t)(kb     ) * HW + row];
      float x1 = zb[(size_t)(kb +  1) * HW + row];
      u0[v] = f2bf(x0) | (f2bf(x1) << 16);
      float y0 = zb[(size_t)(kb + 32) * HW + row];
      float y1 = zb[(size_t)(kb + 33) * HW + row];
      u1[v] = f2bf(y0) | (f2bf(y1) << 16);
    }
    a[m][0] = __builtin_bit_cast(v16bf, u0);
    a[m][1] = __builtin_bit_cast(v16bf, u1);
  }

  float minv[2][8];
  int   mini[2][8];
  #pragma unroll
  for (int m = 0; m < 2; ++m)
    #pragma unroll
    for (int r = 0; r < 8; ++r) { minv[m][r] = 3.4e38f; mini[m][r] = 0; }

  const uint4* bq = reinterpret_cast<const uint4*>(Bpre);
  #define LOADB(JT, U0, U1) do {                                         \
    const uint4* p0 = bq + ((size_t)(((JT) * 2 + 0) * 32 + lane)) * 2;   \
    const uint4* p1 = bq + ((size_t)(((JT) * 2 + 1) * 32 + lane)) * 2;   \
    uint4 q0 = p0[0], q1 = p0[1], q2 = p1[0], q3 = p1[1];                \
    U0 = (v8u){q0.x, q0.y, q0.z, q0.w, q1.x, q1.y, q1.z, q1.w};          \
    U1 = (v8u){q2.x, q2.y, q2.z, q2.w, q3.x, q3.y, q3.z, q3.w};          \
  } while (0)

  v8u cur0, cur1;
  LOADB(0, cur0, cur1);
  float cn = cnorm[lm];

  for (int jt = 0; jt < 32; ++jt) {
    // prefetch next B tile + next cnorm (clamped, branch-free)
    int jn = (jt < 31) ? jt + 1 : jt;
    v8u nxt0, nxt1;
    LOADB(jn, nxt0, nxt1);
    float cn_n = cnorm[jn * 16 + lm];

    v16bf b0 = __builtin_bit_cast(v16bf, cur0);
    v16bf b1 = __builtin_bit_cast(v16bf, cur1);

    v8f acc0 = {0.f, 0.f, 0.f, 0.f, 0.f, 0.f, 0.f, 0.f};
    v8f acc1 = {0.f, 0.f, 0.f, 0.f, 0.f, 0.f, 0.f, 0.f};
    acc0 = wmma_bf16(a[0][0], b0, acc0);
    acc1 = wmma_bf16(a[1][0], b0, acc1);
    acc0 = wmma_bf16(a[0][1], b1, acc0);
    acc1 = wmma_bf16(a[1][1], b1, acc1);

    int j = jt * 16 + lm;          // this lane's column (codebook index)
    #pragma unroll
    for (int r = 0; r < 8; ++r) {  // rows M = 8*h + r
      float d0 = cn - 2.0f * acc0[r];
      if (d0 < minv[0][r]) { minv[0][r] = d0; mini[0][r] = j; }
      float d1 = cn - 2.0f * acc1[r];
      if (d1 < minv[1][r]) { minv[1][r] = d1; mini[1][r] = j; }
    }
    cur0 = nxt0; cur1 = nxt1; cn = cn_n;
  }
  #undef LOADB

  // ---- argmin across the 16 lanes sharing each row (tie -> lower index) ----
  #pragma unroll
  for (int off = 1; off < 16; off <<= 1) {
    #pragma unroll
    for (int m = 0; m < 2; ++m)
      #pragma unroll
      for (int r = 0; r < 8; ++r) {
        float ov = __shfl_xor(minv[m][r], off, 32);
        int   oi = __shfl_xor(mini[m][r], off, 32);
        if (ov < minv[m][r] || (ov == minv[m][r] && oi < mini[m][r])) {
          minv[m][r] = ov; mini[m][r] = oi;
        }
      }
  }
  if (lm == 0) {
    #pragma unroll
    for (int m = 0; m < 2; ++m)
      #pragma unroll
      for (int r = 0; r < 8; ++r) idxS[warp][m][h * 8 + r] = mini[m][r];
  }
  __syncthreads();

  // ---- gather + straight-through output + loss (coalesced along hw) ----
  float* ob = out + ((size_t)b * DIM * HW + hw0);
  float lsum = 0.f;
  #pragma unroll
  for (int m = 0; m < 2; ++m) {
    const int    code = idxS[warp][m][lm];     // row = m*16 + lm
    const float* crow = cb + (size_t)code * DIM;
    #pragma unroll 8
    for (int tt = 0; tt < 32; ++tt) {
      int d = 2 * tt + h;                      // lanes 0-15: even d; 16-31: odd
      float cv = crow[d];
      size_t o2 = (size_t)d * HW + (size_t)(m * 16 + lm);
      float ze = zb[o2];
      float df = ze - cv;
      lsum += df * df;
      ob[o2] = ze + (cv - ze);                 // straight-through, ref fp order
    }
  }

  #pragma unroll
  for (int off = 1; off < 32; off <<= 1) lsum += __shfl_xor(lsum, off, 32);
  if (lane == 0) lossS[warp] = lsum;
  __syncthreads();
  if (tid == 0) {
    float s = 0.f;
    #pragma unroll
    for (int w = 0; w < 8; ++w) s += lossS[w];
    partial[blockIdx.x] = s;
  }
}

// ---------------- deterministic final loss reduction ----------------
__global__ void vq_loss(const float* __restrict__ partial,
                        float* __restrict__ loss_out) {
  __shared__ float sh[64];
  int t = threadIdx.x;                         // 64 threads
  float s = 0.f;
  for (int i = 0; i < NBLOCKS / 64; ++i) s += partial[t * (NBLOCKS / 64) + i];
  sh[t] = s;
  __syncthreads();
  if (t == 0) {
    float tot = 0.f;
    for (int i = 0; i < 64; ++i) tot += sh[i];
    // codebook_loss + BETA*commitment = (1+0.25) * mean(diff^2)
    *loss_out = tot * (1.25f / (float)(BATCH * DIM * HW));
  }
}

extern "C" void kernel_launch(void* const* d_in, const int* in_sizes, int n_in,
                              void* d_out, int out_size, void* d_ws, size_t ws_size,
                              hipStream_t stream) {
  (void)in_sizes; (void)n_in; (void)out_size; (void)ws_size;
  const float* z  = (const float*)d_in[0];
  const float* cb = (const float*)d_in[1];
  float* out = (float*)d_out;

  unsigned int* Bpre    = (unsigned int*)d_ws;            // 16384 u32 = 64 KB
  float*        cnorm   = (float*)d_ws + BPRE_UINTS;      // 512 f32
  float*        partial = cnorm + KENT;                   // 512 f32

  vq_prep<<<64, 256, 0, stream>>>(cb, Bpre, cnorm);
  vq_main<<<NBLOCKS, 256, 0, stream>>>(z, cb, Bpre, cnorm, out, partial);
  vq_loss<<<1, 64, 0, stream>>>(partial, out + (size_t)BATCH * DIM * HW);
}
